// VMD_59047210385694
// MI455X (gfx1250) — matrix operand order repaired and checked by reading, not compiled, not run on hardware
//
#include <hip/hip_runtime.h>
#include <hip/hip_bf16.h>
#include <cstdint>
#include <cstddef>

// ---------------------------------------------------------------------------
// Model constants (fixed by the reference): B,T,X,H,Z,G,Y
// ---------------------------------------------------------------------------
static constexpr int Bc = 4096, Tc = 32, Xc = 256, Hc = 256, Zc = 256, Gc = 128, Yc = 2;

typedef __attribute__((ext_vector_type(16))) __bf16 v16bf;
typedef __attribute__((ext_vector_type(8)))  float  v8f;

__device__ __forceinline__ unsigned short f2bf(float f) {
  unsigned int u = __float_as_uint(f);
  u += 0x7FFFu + ((u >> 16) & 1u);   // round-to-nearest-even
  return (unsigned short)(u >> 16);
}

// tanh via single v_exp_f32: tanh(x) = 1 - 2/(e^{2x}+1); correct limits at +-inf
__device__ __forceinline__ float fast_tanh(float x) {
  float e = __expf(2.f * x);
  return 1.f - 2.f / (e + 1.f);
}

union Frag16 { uint4 q[2]; v16bf v; };

// ---------------------------------------------------------------------------
// Generic bf16 WMMA GEMM: C[M x N] = epilogue(A[M x K] * B[K x N] + bias)
//   A: row-major bf16 (as ushort), lda elements (multiple of 8)
//   Bp: pre-packed bf16 weights; per 16(N)x32(K) tile, per lane, 16 contiguous
//       values matching the v_wmma_f32_16x16x32_bf16 B-operand layout.
//   Block: 256 threads (8 waves). Tile: 128(M) x 64(N). Wave: 16(M) x 64(N).
//   KT (=K/32) is a template constant: full unroll -> immediate-offset loads,
//   and fragments are double-buffered so next-kt loads overlap current WMMAs.
// Epilogues:
//   0: fp32 store        1: sigmoid fp32      2: tanh fp32
//   3: tanh -> bf16      4: fused GRU state update (h = (1-u)h + u*tanh)
// ---------------------------------------------------------------------------
struct GemmArgs {
  const unsigned short* A; int lda;
  const unsigned short* Bp;
  const float* bias;
  float* C; int ldc;
  // epilogue extras
  const float* ru;                        // [M x 512], u at col 256+n
  float* hstate;                          // [M x 256] (read h_prev, write h)
  unsigned short* abuf;                   // [M x 768] bf16, h written at col 256+n
  unsigned short* Cbf; int ldcbf;         // bf16 output for EPI==3
};

template <int EPI, int KT>
__global__ __launch_bounds__(256) void gemm_bf16_wmma(GemmArgs g) {
  const int lane = threadIdx.x & 31;
  const int wv   = threadIdx.x >> 5;
  const int hi   = lane >> 4;       // half-wave select
  const int lc   = lane & 15;
  const int m0   = blockIdx.x * 128 + wv * 16;
  const int ntg  = blockIdx.y * 4;  // first 16-wide n-tile of this block

  v8f acc[4] = {};

  // A-fragment per the ISA 16-bit 16x32 layout:
  //  lanes 0-15: K = {kt*32+0..7, +16..23}; lanes 16-31: K = {+8..15, +24..31}
  const unsigned short* arow = g.A + (size_t)(m0 + lc) * g.lda + hi * 8;
  // B base for this lane / n-group; per (wn,kt) fragment at +(wn*KT+kt)*64 uint4
  const uint4* bbase = (const uint4*)g.Bp + ((size_t)ntg * KT * 32 + lane) * 2;

  Frag16 fa[2];
  Frag16 fb[2][4];

  auto loadA = [&](int kt, Frag16& f) {
    const uint4* pa = (const uint4*)(arow + kt * 32);
    f.q[0] = pa[0];       // 8 bf16 at colbase
    f.q[1] = pa[2];       // 8 bf16 at colbase+16
  };
  auto loadB = [&](int kt, Frag16 (&f)[4]) {
#pragma unroll
    for (int wn = 0; wn < 4; ++wn) {
      const uint4* pb = bbase + (size_t)(wn * KT + kt) * 64;
      f[wn].q[0] = pb[0];
      f[wn].q[1] = pb[1];
    }
  };

  loadA(0, fa[0]);
  loadB(0, fb[0]);
#pragma unroll
  for (int kt = 0; kt < KT; ++kt) {
    const int cur = kt & 1, nxt = cur ^ 1;
    if (kt + 1 < KT) {          // prefetch next-kt fragments before consuming
      loadA(kt + 1, fa[nxt]);
      loadB(kt + 1, fb[nxt]);
    }
#pragma unroll
    for (int wn = 0; wn < 4; ++wn) {
      acc[wn] = __builtin_amdgcn_wmma_f32_16x16x32_bf16(
          false, fa[cur].v, false, fb[cur][wn].v, (short)0, acc[wn], false, false);
    }
  }

  // C layout: acc[wn][r] -> row m0 + r + hi*8, col (ntg+wn)*16 + lc
#pragma unroll
  for (int wn = 0; wn < 4; ++wn) {
    const int n = (ntg + wn) * 16 + lc;
    const float bv = g.bias[n];
#pragma unroll
    for (int r = 0; r < 8; ++r) {
      const int m = m0 + hi * 8 + r;
      float t = acc[wn][r] + bv;
      if (EPI == 0) {
        g.C[(size_t)m * g.ldc + n] = t;
      } else if (EPI == 1) {
        g.C[(size_t)m * g.ldc + n] = 1.f / (1.f + __expf(-t));
      } else if (EPI == 2) {
        g.C[(size_t)m * g.ldc + n] = fast_tanh(t);
      } else if (EPI == 3) {
        g.Cbf[(size_t)m * g.ldcbf + n] = f2bf(fast_tanh(t));
      } else {  // EPI == 4: GRU update
        float u  = g.ru[(size_t)m * 512 + 256 + n];
        float hp = g.hstate[(size_t)m * 256 + n];
        float h  = (1.f - u) * hp + u * fast_tanh(t);
        g.hstate[(size_t)m * 256 + n] = h;
        g.abuf[(size_t)m * 768 + 256 + n] = f2bf(h);
      }
    }
  }
}

// ---------------------------------------------------------------------------
// Weight packing: fp32 [Ksrc x Nsrc] row-major -> packed bf16 fragments.
// dst index = ((nt*KT + kt)*32 + lane)*16 + j ; element = W[k][n] with
//   k = kt*32 + (lane>=16 ? 8 : 0) + (j&7) + (j>=8 ? 16 : 0), n = nt*16 + (lane&15); zero if k>=Ksrc.
// ---------------------------------------------------------------------------
__global__ void k_pack_w(unsigned short* dst, const float* src,
                         int Ksrc, int Nsrc, int KT, int total) {
  int idx = blockIdx.x * blockDim.x + threadIdx.x;
  if (idx >= total) return;
  int j    = idx & 15;
  int lane = (idx >> 4) & 31;
  int tidx = idx >> 9;
  int kt   = tidx % KT;
  int nt   = tidx / KT;
  int k = kt * 32 + (lane >> 4) * 8 + (j & 7) + (j >> 3) * 16;
  int n = nt * 16 + (lane & 15);
  float v = (k < Ksrc) ? src[(size_t)k * Nsrc + n] : 0.f;
  dst[idx] = f2bf(v);
}

__global__ void k_concat2(float* d, const float* a, const float* b, int n) {
  int i = blockIdx.x * blockDim.x + threadIdx.x;
  if (i >= 2 * n) return;
  d[i] = (i < n) ? a[i] : b[i - n];
}

// ---------------------------------------------------------------------------
// Per-step activation prep
// ---------------------------------------------------------------------------
__global__ void k_init_state(float* hstate, float* zstate, unsigned short* Abuf,
                             const float* h0, const float* z0) {
  int i = blockIdx.x * blockDim.x + threadIdx.x;   // Bc*256
  if (i >= Bc * 256) return;
  int b = i >> 8, j = i & 255;
  float h = h0[i], z = z0[i];
  hstate[i] = h;
  zstate[i] = z;
  Abuf[(size_t)b * 768 + 256 + j] = f2bf(h);
  Abuf[(size_t)b * 768 + 512 + j] = f2bf(z);
}

__global__ void k_prep_x(unsigned short* Abuf, const float* x, int t) {
  int i = blockIdx.x * blockDim.x + threadIdx.x;   // Bc*256
  if (i >= Bc * 256) return;
  int b = i >> 8, j = i & 255;
  Abuf[(size_t)b * 768 + j] = f2bf(x[((size_t)b * Tc + t) * Xc + j]);
}

__global__ void k_prep_a2(unsigned short* A2, const float* x, const float* ru,
                          const float* hstate, const float* zstate, int t) {
  int i = blockIdx.x * blockDim.x + threadIdx.x;   // Bc*768
  if (i >= Bc * 768) return;
  int b = i / 768, j = i - b * 768;
  float v;
  if (j < 256)      v = x[((size_t)b * Tc + t) * Xc + j];
  else if (j < 512) v = ru[(size_t)b * 512 + (j - 256)] * hstate[(size_t)b * 256 + (j - 256)];
  else              v = zstate[(size_t)b * 256 + (j - 512)];
  A2[(size_t)b * 768 + j] = f2bf(v);
}

__global__ void k_prep_a3(unsigned short* A3, const float* x, const float* hstate,
                          const float* yph, int t) {
  int i = blockIdx.x * blockDim.x + threadIdx.x;   // Bc*544
  if (i >= Bc * 544) return;
  int b = i / 544, j = i - b * 544;
  float v;
  if (j < 256)      v = x[((size_t)b * Tc + t) * Xc + j];
  else if (j < 512) v = hstate[(size_t)b * 256 + (j - 256)];
  else if (j < 514) v = yph[((size_t)b * Tc + t) * Yc + (j - 512)];
  else              v = 0.f;
  A3[(size_t)b * 544 + j] = f2bf(v);
}

// ---------------------------------------------------------------------------
// Reparameterized z + KL row-reduction. One block (256 threads) per batch row.
// ---------------------------------------------------------------------------
__global__ void k_finalize(const float* pstats, const float* qstats, const float* eps_t,
                           float* zstate, unsigned short* Abuf, float* kl_out, int t) {
  int b = blockIdx.x, z = threadIdx.x;
  float mp  = pstats[(size_t)b * 512 + z];
  float lvp = pstats[(size_t)b * 512 + 256 + z];
  float mq  = qstats[(size_t)b * 512 + z];
  float lvq = qstats[(size_t)b * 512 + 256 + z];
  float e   = eps_t[(size_t)b * 256 + z];
  float sdq = __expf(0.5f * lvq);
  float zp  = mq + sdq * e;
  zstate[(size_t)b * 256 + z] = zp;
  Abuf[(size_t)b * 768 + 512 + z] = f2bf(zp);
  float d   = mq - mp;
  float kle = 0.5f * (lvp - lvq) + (__expf(lvq) + d * d) * 0.5f * __expf(-lvp) - 0.5f;
  __shared__ float red[256];
  red[z] = kle;
  __syncthreads();
  for (int s = 128; s > 0; s >>= 1) {
    if (z < s) red[z] += red[z + s];
    __syncthreads();
  }
  if (z == 0) kl_out[(size_t)b * Tc + t] = red[0];
}

// ---------------------------------------------------------------------------
// Final head: y = softmax(g @ Wy + by) over Y=2; one thread per (b,t) row.
// ---------------------------------------------------------------------------
__global__ void k_y_softmax(const float* gbuf, const float* Wy, const float* by,
                            float* y_out) {
  int i = blockIdx.x * blockDim.x + threadIdx.x;   // Bc*Tc rows
  if (i >= Bc * Tc) return;
  const float4* gr = (const float4*)(gbuf + (size_t)i * Gc);
  float a0 = by[0], a1 = by[1];
#pragma unroll 8
  for (int k = 0; k < Gc / 4; ++k) {
    float4 v = gr[k];
    int kb = 4 * k;
    a0 += v.x * Wy[2 * (kb + 0)] + v.y * Wy[2 * (kb + 1)] + v.z * Wy[2 * (kb + 2)] + v.w * Wy[2 * (kb + 3)];
    a1 += v.x * Wy[2 * (kb + 0) + 1] + v.y * Wy[2 * (kb + 1) + 1] + v.z * Wy[2 * (kb + 2) + 1] + v.w * Wy[2 * (kb + 3) + 1];
  }
  float m = fmaxf(a0, a1);
  float e0 = __expf(a0 - m), e1 = __expf(a1 - m);
  float s = e0 + e1;
  y_out[2 * (size_t)i]     = e0 / s;
  y_out[2 * (size_t)i + 1] = e1 / s;
}

__global__ void k_gather_gT(const float* gbuf, const int* Tph, float* gT) {
  int i = blockIdx.x * blockDim.x + threadIdx.x;   // Bc*Gc
  if (i >= Bc * Gc) return;
  int b = i >> 7, j = i & 127;
  int tt = Tph[b] - 1;
  tt = tt < 0 ? 0 : (tt > Tc - 1 ? Tc - 1 : tt);
  gT[i] = gbuf[((size_t)b * Tc + tt) * Gc + j];
}

// ---------------------------------------------------------------------------
// Host driver
// ---------------------------------------------------------------------------
extern "C" void kernel_launch(void* const* d_in, const int* in_sizes, int n_in,
                              void* d_out, int out_size, void* d_ws, size_t ws_size,
                              hipStream_t stream) {
  (void)in_sizes; (void)n_in; (void)out_size; (void)ws_size;
  const float* x    = (const float*)d_in[0];
  const float* y_ph = (const float*)d_in[1];
  const int*   Tph  = (const int*)d_in[2];
  const float* h0   = (const float*)d_in[3];
  const float* z0   = (const float*)d_in[4];
  const float* eps  = (const float*)d_in[5];
  const float* Wr  = (const float*)d_in[6];  const float* br  = (const float*)d_in[7];
  const float* Wu  = (const float*)d_in[8];  const float* bu  = (const float*)d_in[9];
  const float* Wh  = (const float*)d_in[10]; const float* bh  = (const float*)d_in[11];
  const float* Wzp = (const float*)d_in[12]; const float* bzp = (const float*)d_in[13];
  const float* Wpm = (const float*)d_in[14]; const float* bpm = (const float*)d_in[15];
  const float* Wps = (const float*)d_in[16]; const float* bps = (const float*)d_in[17];
  const float* Wzq = (const float*)d_in[18]; const float* bzq = (const float*)d_in[19];
  const float* Wqm = (const float*)d_in[20]; const float* bqm = (const float*)d_in[21];
  const float* Wqs = (const float*)d_in[22]; const float* bqs = (const float*)d_in[23];
  const float* Wg  = (const float*)d_in[24]; const float* bg  = (const float*)d_in[25];
  const float* Wy  = (const float*)d_in[26]; const float* by  = (const float*)d_in[27];

  float* outy  = (float*)d_out;                      // [B,T,Y]
  float* outkl = outy + (size_t)Bc * Tc * Yc;        // [B,T]
  float* outgT = outkl + (size_t)Bc * Tc;            // [B,G]

  // --- workspace carve (bump allocator, 256B aligned, deterministic) ---
  char* p = (char*)d_ws;
  auto take = [&](size_t bytes) { char* r = p; p += (bytes + 255) & ~(size_t)255; return r; };

  constexpr int KT768 = 24, KT512 = 16, KT544 = 17, KT256 = 8;
  unsigned short* pk_ru = (unsigned short*)take((size_t)768 * 512 * 2); // [Wr|Wu]
  unsigned short* pk_h  = (unsigned short*)take((size_t)768 * 256 * 2);
  unsigned short* pk_zp = (unsigned short*)take((size_t)512 * 256 * 2);
  unsigned short* pk_zq = (unsigned short*)take((size_t)544 * 256 * 2); // K padded 514->544
  unsigned short* pk_pm = (unsigned short*)take((size_t)256 * 512 * 2); // [Wpm|Wps]
  unsigned short* pk_qm = (unsigned short*)take((size_t)256 * 512 * 2); // [Wqm|Wqs]
  unsigned short* pk_g  = (unsigned short*)take((size_t)768 * 128 * 2);
  float* b_ru   = (float*)take(512 * 4);
  float* b_pmps = (float*)take(512 * 4);
  float* b_qmqs = (float*)take(512 * 4);
  unsigned short* Abuf = (unsigned short*)take((size_t)Bc * 768 * 2);  // [x|h|z] bf16
  unsigned short* A2   = (unsigned short*)take((size_t)Bc * 768 * 2);  // [x|r*h|z]
  unsigned short* A3   = (unsigned short*)take((size_t)Bc * 544 * 2);  // [x|h|y|pad]
  unsigned short* hzp  = (unsigned short*)take((size_t)Bc * 256 * 2);
  unsigned short* hzq  = (unsigned short*)take((size_t)Bc * 256 * 2);
  float* ru     = (float*)take((size_t)Bc * 512 * 4);   // [r|u]
  float* pstats = (float*)take((size_t)Bc * 512 * 4);   // [mu_p|logvar_p]
  float* qstats = (float*)take((size_t)Bc * 512 * 4);   // [mu_q|logvar_q]
  float* hstate = (float*)take((size_t)Bc * 256 * 4);
  float* zstate = (float*)take((size_t)Bc * 256 * 4);
  float* gbuf   = (float*)take((size_t)Bc * Tc * Gc * 4);

  // --- one-time weight packing / bias concat ---
  auto packW = [&](unsigned short* dst, const float* src, int Ksrc, int Nsrc, int KT) {
    int total = (Nsrc / 16) * KT * 512;
    k_pack_w<<<dim3((total + 255) / 256), dim3(256), 0, stream>>>(dst, src, Ksrc, Nsrc, KT, total);
  };
  packW(pk_ru,                      Wr, 768, 256, KT768);
  packW(pk_ru + 16 * KT768 * 512,   Wu, 768, 256, KT768);
  packW(pk_h,                       Wh, 768, 256, KT768);
  packW(pk_zp,                      Wzp, 512, 256, KT512);
  packW(pk_zq,                      Wzq, 514, 256, KT544);
  packW(pk_pm,                      Wpm, 256, 256, KT256);
  packW(pk_pm + 16 * KT256 * 512,   Wps, 256, 256, KT256);
  packW(pk_qm,                      Wqm, 256, 256, KT256);
  packW(pk_qm + 16 * KT256 * 512,   Wqs, 256, 256, KT256);
  packW(pk_g,                       Wg, 768, 128, KT768);
  k_concat2<<<dim3(4), dim3(256), 0, stream>>>(b_ru,   br,  bu,  256);
  k_concat2<<<dim3(4), dim3(256), 0, stream>>>(b_pmps, bpm, bps, 256);
  k_concat2<<<dim3(4), dim3(256), 0, stream>>>(b_qmqs, bqm, bqs, 256);

  // --- init recurrent state ---
  k_init_state<<<dim3(Bc * 256 / 256), dim3(256), 0, stream>>>(hstate, zstate, Abuf, h0, z0);

  const dim3 blk(256);
  // --- scan over time ---
  for (int t = 0; t < Tc; ++t) {
    // x_t into Abuf cols [0,256)
    k_prep_x<<<dim3(Bc * 256 / 256), blk, 0, stream>>>(Abuf, x, t);

    // gates: [x,h,z] @ [Wr|Wu], sigmoid
    { GemmArgs a{}; a.A = Abuf; a.lda = 768; a.Bp = pk_ru;
      a.bias = b_ru; a.C = ru; a.ldc = 512;
      gemm_bf16_wmma<1, KT768><<<dim3(32, 8), blk, 0, stream>>>(a); }

    // candidate input: [x, r*h, z]
    k_prep_a2<<<dim3(Bc * 768 / 256), blk, 0, stream>>>(A2, x, ru, hstate, zstate, t);

    // h_tilde GEMM + fused GRU update (writes hstate + bf16 h into Abuf)
    { GemmArgs a{}; a.A = A2; a.lda = 768; a.Bp = pk_h;
      a.bias = bh; a.ru = ru; a.hstate = hstate; a.abuf = Abuf;
      gemm_bf16_wmma<4, KT768><<<dim3(32, 4), blk, 0, stream>>>(a); }

    // [x, h, y, pad] for latent nets
    k_prep_a3<<<dim3((Bc * 544 + 255) / 256), blk, 0, stream>>>(A3, x, hstate, y_ph, t);

    // prior hidden: tanh([x,h] @ Wzp) -> bf16
    { GemmArgs a{}; a.A = A3; a.lda = 544; a.Bp = pk_zp;
      a.bias = bzp; a.Cbf = hzp; a.ldcbf = 256;
      gemm_bf16_wmma<3, KT512><<<dim3(32, 4), blk, 0, stream>>>(a); }

    // posterior hidden: tanh([x,h,y,pad] @ Wzq_pad) -> bf16
    { GemmArgs a{}; a.A = A3; a.lda = 544; a.Bp = pk_zq;
      a.bias = bzq; a.Cbf = hzq; a.ldcbf = 256;
      gemm_bf16_wmma<3, KT544><<<dim3(32, 4), blk, 0, stream>>>(a); }

    // prior stats: hzp @ [Wpm|Wps]
    { GemmArgs a{}; a.A = hzp; a.lda = 256; a.Bp = pk_pm;
      a.bias = b_pmps; a.C = pstats; a.ldc = 512;
      gemm_bf16_wmma<0, KT256><<<dim3(32, 8), blk, 0, stream>>>(a); }

    // posterior stats: hzq @ [Wqm|Wqs]
    { GemmArgs a{}; a.A = hzq; a.lda = 256; a.Bp = pk_qm;
      a.bias = b_qmqs; a.C = qstats; a.ldc = 512;
      gemm_bf16_wmma<0, KT256><<<dim3(32, 8), blk, 0, stream>>>(a); }

    // z sample + KL (writes zstate, bf16 z into Abuf, kl[b,t])
    k_finalize<<<dim3(Bc), blk, 0, stream>>>(pstats, qstats,
        eps + (size_t)t * Bc * Zc, zstate, Abuf, outkl, t);

    // g_t = tanh([x,h,z] @ Wg) into gbuf[:, t, :]
    { GemmArgs a{}; a.A = Abuf; a.lda = 768; a.Bp = pk_g;
      a.bias = bg; a.C = gbuf + (size_t)t * Gc; a.ldc = Tc * Gc;
      gemm_bf16_wmma<2, KT768><<<dim3(32, 2), blk, 0, stream>>>(a); }
  }

  // --- output head + gather ---
  k_y_softmax<<<dim3(Bc * Tc / 256), blk, 0, stream>>>(gbuf, Wy, by, outy);
  k_gather_gT<<<dim3(Bc * Gc / 256), blk, 0, stream>>>(gbuf, Tph, outgT);
}
